// SmallEncoder_47854525612174
// MI455X (gfx1250) — compile-verified, matrix operand order checked by reading, not run on hardware
//
#include <hip/hip_runtime.h>
#include <hip/hip_bf16.h>

typedef __attribute__((ext_vector_type(16))) _Float16 v16h;
typedef __attribute__((ext_vector_type(8)))  float    v8f;

static inline int CDIV(size_t a, int b) { return (int)((a + (size_t)b - 1) / (size_t)b); }

// ---------------- elementwise / utility kernels ----------------

__global__ void zero_kernel(float* __restrict__ p, size_t n) {
  size_t i = (size_t)blockIdx.x * blockDim.x + threadIdx.x;
  if (i < n) p[i] = 0.f;
}

__global__ void copy_kernel(float* __restrict__ d, const float* __restrict__ s, size_t n) {
  size_t i = (size_t)blockIdx.x * blockDim.x + threadIdx.x;
  if (i < n) d[i] = s[i];
}

// x:[B,N,3] -> xp:[B,N,16] zero padded
__global__ void pad_in_kernel(float* __restrict__ xp, const float* __restrict__ x, int B, int N) {
  size_t i = (size_t)blockIdx.x * blockDim.x + threadIdx.x;
  size_t tot = (size_t)B * N * 16;
  if (i >= tot) return;
  int c = (int)(i & 15);
  size_t bn = i >> 4;
  xp[i] = (c < 3) ? x[bn * 3 + c] : 0.f;
}

// out[b,dst[e],c] += w[e]*x[b,src[e],c]  (COO scatter, fp32 atomics)
__global__ void prop_kernel(float* __restrict__ out, const float* __restrict__ x,
                            const int* __restrict__ src, const int* __restrict__ dst,
                            const float* __restrict__ w, int E, int N, int C, int Cpad, int B) {
  int e = blockIdx.x * blockDim.x + threadIdx.x;
  if (e >= E) return;
  int s = src[e], d = dst[e];
  float we = w[e];
  for (int b = 0; b < B; ++b) {
    const float* xs = x + ((size_t)b * N + s) * Cpad;
    float* od = out + ((size_t)b * N + d) * Cpad;
    for (int c = 0; c < C; ++c)
      (void)__hip_atomic_fetch_add(&od[c], we * xs[c], __ATOMIC_RELAXED, __HIP_MEMORY_SCOPE_AGENT);
  }
}

// t2 = 2*p - x   (in-place over p allowed)
__global__ void cheb2_kernel(float* __restrict__ t2, const float* __restrict__ p,
                             const float* __restrict__ x, size_t n) {
  size_t i = (size_t)blockIdx.x * blockDim.x + threadIdx.x;
  if (i < n) t2[i] = 2.f * p[i] - x[i];
}

// out = relu(a + b)
__global__ void add_relu_kernel(float* __restrict__ out, const float* __restrict__ a,
                                const float* __restrict__ b, size_t n) {
  size_t i = (size_t)blockIdx.x * blockDim.x + threadIdx.x;
  if (i < n) out[i] = fmaxf(a[i] + b[i], 0.f);
}

// max-pool 4 consecutive nodes
__global__ void pool_kernel(float* __restrict__ out, const float* __restrict__ in,
                            int B, int Nout, int Cpad) {
  size_t i = (size_t)blockIdx.x * blockDim.x + threadIdx.x;
  size_t tot = (size_t)B * Nout * Cpad;
  if (i >= tot) return;
  int c = (int)(i % Cpad);
  size_t bn = i / Cpad;
  int b = (int)(bn / Nout);
  int np = (int)(bn % Nout);
  const float* base = in + (((size_t)b * Nout * 4) + (size_t)np * 4) * Cpad + c;
  float m = base[0];
  m = fmaxf(m, base[Cpad]);
  m = fmaxf(m, base[2 * (size_t)Cpad]);
  m = fmaxf(m, base[3 * (size_t)Cpad]);
  out[i] = m;
}

// ---------------- GroupNorm (G=8) ----------------

__global__ void gn_reduce_kernel(const float* __restrict__ x, float* __restrict__ stats,
                                 int B, int N, int C, int Cpad, int nblk) {
  int bg = blockIdx.x / nblk;
  int part = blockIdx.x % nblk;
  int b = bg >> 3, g = bg & 7;
  int cpg = C >> 3;
  size_t total = (size_t)N * cpg;
  float s = 0.f, ss = 0.f;
  for (size_t i = (size_t)part * blockDim.x + threadIdx.x; i < total;
       i += (size_t)nblk * blockDim.x) {
    int n = (int)(i / cpg), j = (int)(i % cpg);
    float v = x[((size_t)b * N + n) * Cpad + g * cpg + j];
    s += v; ss += v * v;
  }
  __shared__ float sh0[256];
  __shared__ float sh1[256];
  sh0[threadIdx.x] = s; sh1[threadIdx.x] = ss;
  __syncthreads();
  for (int o = 128; o > 0; o >>= 1) {
    if ((int)threadIdx.x < o) { sh0[threadIdx.x] += sh0[threadIdx.x + o]; sh1[threadIdx.x] += sh1[threadIdx.x + o]; }
    __syncthreads();
  }
  if (threadIdx.x == 0) {
    (void)__hip_atomic_fetch_add(&stats[bg * 2],     sh0[0], __ATOMIC_RELAXED, __HIP_MEMORY_SCOPE_AGENT);
    (void)__hip_atomic_fetch_add(&stats[bg * 2 + 1], sh1[0], __ATOMIC_RELAXED, __HIP_MEMORY_SCOPE_AGENT);
  }
}

__global__ void gn_apply_kernel(float* __restrict__ y, const float* __restrict__ x,
                                const float* __restrict__ stats,
                                const float* __restrict__ gamma, const float* __restrict__ beta,
                                int B, int N, int C, int Cpad, int dorelu) {
  size_t i = (size_t)blockIdx.x * blockDim.x + threadIdx.x;
  size_t tot = (size_t)B * N * Cpad;
  if (i >= tot) return;
  int c = (int)(i % Cpad);
  if (c >= C) { y[i] = 0.f; return; }
  int b = (int)((i / Cpad) / N);
  int cpg = C >> 3;
  int g = c / cpg;
  float cnt = (float)N * (float)cpg;
  float mu = stats[(b * 8 + g) * 2] / cnt;
  float var = stats[(b * 8 + g) * 2 + 1] / cnt - mu * mu;
  float v = (x[i] - mu) * rsqrtf(var + 1e-5f) * gamma[c] + beta[c];
  y[i] = dorelu ? fmaxf(v, 0.f) : v;
}

// ---------------- theta pre-pack into WMMA B-register layout ----------------
// pk[((ct*KB + kb)*32 + lane)*16 + i] = f16(theta[k, co]) with zero padding
//   i<8 : k = kb*32 + (lane>=16)*8 + i
//   i>=8: k = kb*32 + 16 + (lane>=16)*8 + (i-8)
//   co  = ct*16 + (lane&15)
__global__ void pack_theta_kernel(_Float16* __restrict__ pk, const float* __restrict__ th,
                                  int Cin, int Cout, int KB, int CT) {
  int idx = blockIdx.x * blockDim.x + threadIdx.x;
  int tot = CT * KB * 32 * 16;
  if (idx >= tot) return;
  int i = idx & 15;
  int lane = (idx >> 4) & 31;
  int blk = idx >> 9;           // ct*KB + kb
  int kb = blk % KB;
  int ct = blk / KB;
  int hl = lane >> 4, lan = lane & 15;
  int k = (i < 8) ? (kb * 32 + hl * 8 + i) : (kb * 32 + 16 + hl * 8 + (i - 8));
  int co = ct * 16 + lan;
  pk[idx] = (k < Cin && co < Cout) ? (_Float16)th[(size_t)k * Cout + co] : (_Float16)0.f;
}

// ---------------- WMMA GEMM: out[M,CoutPad] (+)= x[M,CinPad] @ theta ----------------
// 256 threads = 8 waves; wave w handles row tile m0=(blockIdx.x*8+w)*16, all CT col tiles.
// HALFK=1 <=> CinPad==16 (upper 16 halves of A are compile-time zero).
template <int HALFK, int CT>
__global__ __launch_bounds__(256)
void wmma_gemm_kernel(float* __restrict__ out, const float* __restrict__ x,
                      const _Float16* __restrict__ bpk, const float* __restrict__ bias,
                      int CinPad, int Cout, int CoutPad, int KB, int accumulate) {
  const int lane = threadIdx.x & 31;
  const int wv = threadIdx.x >> 5;
  const int hl = lane >> 4;
  const int lan = lane & 15;
  const int m0 = (blockIdx.x * 8 + wv) * 16;
  const size_t arow = (size_t)(m0 + lan) * CinPad;

  v8f acc[CT];
#pragma unroll
  for (int ct = 0; ct < CT; ++ct) {
    if (accumulate) {
#pragma unroll
      for (int r = 0; r < 8; ++r)
        acc[ct][r] = out[(size_t)(m0 + r + 8 * hl) * CoutPad + ct * 16 + lan];
    } else {
      int co = ct * 16 + lan;
      float bv = (bias != nullptr && co < Cout) ? bias[co] : 0.f;
#pragma unroll
      for (int r = 0; r < 8; ++r) acc[ct][r] = bv;
    }
  }

  for (int kb = 0; kb < KB; ++kb) {
    const int kc = kb * 32;
    v16h a;
    float4 a0 = *(const float4*)(x + arow + kc + hl * 8);
    float4 a1 = *(const float4*)(x + arow + kc + hl * 8 + 4);
    a[0] = (_Float16)a0.x; a[1] = (_Float16)a0.y; a[2] = (_Float16)a0.z; a[3] = (_Float16)a0.w;
    a[4] = (_Float16)a1.x; a[5] = (_Float16)a1.y; a[6] = (_Float16)a1.z; a[7] = (_Float16)a1.w;
    if (HALFK) {
#pragma unroll
      for (int j = 8; j < 16; ++j) a[j] = (_Float16)0.f;
    } else {
      float4 a2 = *(const float4*)(x + arow + kc + 16 + hl * 8);
      float4 a3 = *(const float4*)(x + arow + kc + 16 + hl * 8 + 4);
      a[8]  = (_Float16)a2.x; a[9]  = (_Float16)a2.y; a[10] = (_Float16)a2.z; a[11] = (_Float16)a2.w;
      a[12] = (_Float16)a3.x; a[13] = (_Float16)a3.y; a[14] = (_Float16)a3.z; a[15] = (_Float16)a3.w;
    }
#pragma unroll
    for (int ct = 0; ct < CT; ++ct) {
      v16h b = *(const v16h*)(bpk + ((size_t)(ct * KB + kb) * 32 + lane) * 16);
      acc[ct] = __builtin_amdgcn_wmma_f32_16x16x32_f16(false, a, false, b, (short)0, acc[ct], false, false);
    }
  }

#pragma unroll
  for (int ct = 0; ct < CT; ++ct)
#pragma unroll
    for (int r = 0; r < 8; ++r)
      out[(size_t)(m0 + r + 8 * hl) * CoutPad + ct * 16 + lan] = acc[ct][r];
}

// ---------------- host-side orchestration ----------------

struct Graph { const int* src; const int* dst; const float* w; int E; };

static void zero(float* p, size_t n, hipStream_t s) {
  zero_kernel<<<CDIV(n, 256), 256, 0, s>>>(p, n);
}

static void gemm(float* out, const float* x, const float* th, const float* bias,
                 int M, int Cin, int CinPad, int Cout, int CoutPad, int acc,
                 _Float16* pk, hipStream_t s) {
  const int KB = (CinPad >= 32) ? (CinPad / 32) : 1;
  const int HALFK = (CinPad == 16) ? 1 : 0;
  const int CT = CoutPad / 16;
  int tot = CT * KB * 32 * 16;
  pack_theta_kernel<<<CDIV((size_t)tot, 256), 256, 0, s>>>(pk, th, Cin, Cout, KB, CT);
  dim3 grid(M / 128, 1, 1);
#define GEMM_CASE(H, C) \
  wmma_gemm_kernel<H, C><<<grid, 256, 0, s>>>(out, x, pk, bias, CinPad, Cout, CoutPad, KB, acc)
  if (HALFK) {
    switch (CT) {
      case 1: GEMM_CASE(1, 1); break;
      case 2: GEMM_CASE(1, 2); break;
      case 4: GEMM_CASE(1, 4); break;
      case 6: GEMM_CASE(1, 6); break;
      default: GEMM_CASE(1, 8); break;
    }
  } else {
    switch (CT) {
      case 1: GEMM_CASE(0, 1); break;
      case 2: GEMM_CASE(0, 2); break;
      case 4: GEMM_CASE(0, 4); break;
      case 6: GEMM_CASE(0, 6); break;
      default: GEMM_CASE(0, 8); break;
    }
  }
#undef GEMM_CASE
}

// out = sum_k T_k(x) @ theta_k + bias ; K in {1,3}; t1,t2 sized [B*N, CinPad]
static void cheb(float* out, const float* x, const float* th, const float* bias, int K,
                 int B, int N, int Cin, int CinPad, int Cout, int CoutPad,
                 const Graph& g, float* t1, float* t2, _Float16* pk, hipStream_t s) {
  int M = B * N;
  gemm(out, x, th, bias, M, Cin, CinPad, Cout, CoutPad, 0, pk, s);
  if (K >= 3) {
    size_t tn = (size_t)M * CinPad;
    zero(t1, tn, s);
    prop_kernel<<<CDIV((size_t)g.E, 256), 256, 0, s>>>(t1, x, g.src, g.dst, g.w, g.E, N, Cin, CinPad, B);
    gemm(out, t1, th + (size_t)Cin * Cout, nullptr, M, Cin, CinPad, Cout, CoutPad, 1, pk, s);
    zero(t2, tn, s);
    prop_kernel<<<CDIV((size_t)g.E, 256), 256, 0, s>>>(t2, t1, g.src, g.dst, g.w, g.E, N, Cin, CinPad, B);
    cheb2_kernel<<<CDIV(tn, 256), 256, 0, s>>>(t2, t2, x, tn);
    gemm(out, t2, th + 2 * (size_t)Cin * Cout, nullptr, M, Cin, CinPad, Cout, CoutPad, 1, pk, s);
  }
}

static void gnorm(float* y, const float* x, const float* gamma, const float* beta,
                  int B, int N, int C, int Cpad, int relu, float* stats, hipStream_t s) {
  const int NBLK = 64;
  zero(stats, (size_t)B * 16, s);
  gn_reduce_kernel<<<B * 8 * NBLK, 256, 0, s>>>(x, stats, B, N, C, Cpad, NBLK);
  size_t tot = (size_t)B * N * Cpad;
  gn_apply_kernel<<<CDIV(tot, 256), 256, 0, s>>>(y, x, stats, gamma, beta, B, N, C, Cpad, relu);
}

// leaves (jax dict-sorted order):
// no shortcut: c1_b,c1_t,c2_b,c2_t,c3_b,c3_t,n1_b,n1_g,n2_b,n2_g,n3_b,n3_g
// shortcut:    ...,c3_t,cs_b,cs_t,n1_b,...,n3_g,ns_b,ns_g
static void run_block(void* const* lv, bool sc, int B, int N,
                      int Cin, int CinPad, int Cout, int CoutPad, int P4, int P4Pad,
                      const Graph& g, float* X, float* Y, float* U, float* V,
                      float* T1, float* T2, float* SC, float* stats, _Float16* pk,
                      hipStream_t s) {
  const float* c1_b = (const float*)lv[0];  const float* c1_t = (const float*)lv[1];
  const float* c2_b = (const float*)lv[2];  const float* c2_t = (const float*)lv[3];
  const float* c3_b = (const float*)lv[4];  const float* c3_t = (const float*)lv[5];
  int o = sc ? 8 : 6;
  const float* cs_b = sc ? (const float*)lv[6] : nullptr;
  const float* cs_t = sc ? (const float*)lv[7] : nullptr;
  const float* n1_b = (const float*)lv[o + 0]; const float* n1_g = (const float*)lv[o + 1];
  const float* n2_b = (const float*)lv[o + 2]; const float* n2_g = (const float*)lv[o + 3];
  const float* n3_b = (const float*)lv[o + 4]; const float* n3_g = (const float*)lv[o + 5];
  const float* ns_b = sc ? (const float*)lv[o + 6] : nullptr;
  const float* ns_g = sc ? (const float*)lv[o + 7] : nullptr;

  cheb(U, X, c1_t, c1_b, 1, B, N, Cin, CinPad, P4, P4Pad, g, T1, T2, pk, s);
  gnorm(U, U, n1_g, n1_b, B, N, P4, P4Pad, 1, stats, s);
  cheb(V, U, c2_t, c2_b, 3, B, N, P4, P4Pad, P4, P4Pad, g, T1, T2, pk, s);
  gnorm(V, V, n2_g, n2_b, B, N, P4, P4Pad, 1, stats, s);
  cheb(Y, V, c3_t, c3_b, 1, B, N, P4, P4Pad, Cout, CoutPad, g, T1, T2, pk, s);
  gnorm(Y, Y, n3_g, n3_b, B, N, Cout, CoutPad, 1, stats, s);
  size_t tot = (size_t)B * N * CoutPad;
  if (sc) {
    cheb(SC, X, cs_t, cs_b, 1, B, N, Cin, CinPad, Cout, CoutPad, g, T1, T2, pk, s);
    gnorm(SC, SC, ns_g, ns_b, B, N, Cout, CoutPad, 0, stats, s);
    add_relu_kernel<<<CDIV(tot, 256), 256, 0, s>>>(X, SC, Y, tot);
  } else {
    add_relu_kernel<<<CDIV(tot, 256), 256, 0, s>>>(X, X, Y, tot);
  }
}

extern "C" void kernel_launch(void* const* d_in, const int* in_sizes, int n_in,
                              void* d_out, int out_size, void* d_ws, size_t ws_size,
                              hipStream_t stream) {
  const int B = 2, N0 = 196608, N1 = N0 / 4, N2 = N0 / 16, N3 = N0 / 64;
  const float* x_in = (const float*)d_in[0];
  Graph g0{(const int*)d_in[1],  (const int*)d_in[2],  (const float*)d_in[3],  in_sizes[1]};
  Graph g1{(const int*)d_in[4],  (const int*)d_in[5],  (const float*)d_in[6],  in_sizes[4]};
  Graph g2{(const int*)d_in[7],  (const int*)d_in[8],  (const float*)d_in[9],  in_sizes[7]};
  Graph g3{(const int*)d_in[10], (const int*)d_in[11], (const float*)d_in[12], in_sizes[10]};

  // params flattened in jax tree order (dicts sorted by key), appended after the 13 tensors:
  // 13 conv1_bias, 14 conv1_theta, 15 conv2_bias, 16 conv2_theta,
  // 17..28 l1b1, 29..40 l1b2, 41..56 l2b1, 57..68 l2b2, 69..84 l3b1, 85..96 l3b2,
  // 97 norm1_b, 98 norm1_g
  const float* conv1_b = (const float*)d_in[13];
  const float* conv1_t = (const float*)d_in[14];
  const float* conv2_b = (const float*)d_in[15];
  const float* conv2_t = (const float*)d_in[16];
  const float* norm1_b = (const float*)d_in[97];
  const float* norm1_g = (const float*)d_in[98];

  // workspace arena (~227 MB)
  char* wp = (char*)d_ws;
  auto alloc = [&](size_t bytes) {
    float* p = (float*)wp;
    wp += (bytes + 255) & ~(size_t)255;
    return p;
  };
  float* X  = alloc((size_t)B * N0 * 32 * 4);   // current activation (max 50MB)
  float* Y  = alloc((size_t)B * N0 * 32 * 4);   // block output / pool staging
  float* U  = alloc((size_t)B * N0 * 16 * 4);   // p4 intermediate 1
  float* V  = alloc((size_t)B * N0 * 16 * 4);   // p4 intermediate 2
  float* T1 = alloc((size_t)B * N0 * 16 * 4);   // prop temp
  float* T2 = alloc((size_t)B * N0 * 16 * 4);   // prop temp
  float* SC = alloc((size_t)B * N1 * 64 * 4);   // shortcut / padded input (reused)
  float* stats = alloc(256);
  _Float16* PK = (_Float16*)alloc(32768);       // packed theta (<=24KB)
  float* XP = SC;                               // padded input x: [B,N0,16]

  // --- stem: pad input, conv1 (K=3, 3->32), GN+ReLU ---
  {
    size_t tot = (size_t)B * N0 * 16;
    pad_in_kernel<<<CDIV(tot, 256), 256, 0, stream>>>(XP, x_in, B, N0);
  }
  cheb(X, XP, conv1_t, conv1_b, 3, B, N0, 3, 16, 32, 32, g0, T1, T2, PK, stream);
  gnorm(X, X, norm1_g, norm1_b, B, N0, 32, 32, 1, stats, stream);

  // --- layer 1: two blocks 32->32 (p4=8 pad16), pool ---
  run_block(&d_in[17], false, B, N0, 32, 32, 32, 32, 8, 16, g0, X, Y, U, V, T1, T2, SC, stats, PK, stream);
  run_block(&d_in[29], false, B, N0, 32, 32, 32, 32, 8, 16, g0, X, Y, U, V, T1, T2, SC, stats, PK, stream);
  {
    size_t tot = (size_t)B * N1 * 32;
    pool_kernel<<<CDIV(tot, 256), 256, 0, stream>>>(Y, X, B, N1, 32);
    copy_kernel<<<CDIV(tot, 256), 256, 0, stream>>>(X, Y, tot);
  }

  // --- layer 2: 32->64 (shortcut), 64->64 ; p4=16 ; pool ---
  run_block(&d_in[41], true,  B, N1, 32, 32, 64, 64, 16, 16, g1, X, Y, U, V, T1, T2, SC, stats, PK, stream);
  run_block(&d_in[57], false, B, N1, 64, 64, 64, 64, 16, 16, g1, X, Y, U, V, T1, T2, SC, stats, PK, stream);
  {
    size_t tot = (size_t)B * N2 * 64;
    pool_kernel<<<CDIV(tot, 256), 256, 0, stream>>>(Y, X, B, N2, 64);
    copy_kernel<<<CDIV(tot, 256), 256, 0, stream>>>(X, Y, tot);
  }

  // --- layer 3: 64->96 (shortcut), 96->96 ; p4=24 pad32 ; pool ---
  run_block(&d_in[69], true,  B, N2, 64, 64, 96, 96, 24, 32, g2, X, Y, U, V, T1, T2, SC, stats, PK, stream);
  run_block(&d_in[85], false, B, N2, 96, 96, 96, 96, 24, 32, g2, X, Y, U, V, T1, T2, SC, stats, PK, stream);
  {
    size_t tot = (size_t)B * N3 * 96;
    pool_kernel<<<CDIV(tot, 256), 256, 0, stream>>>(Y, X, B, N3, 96);
    copy_kernel<<<CDIV(tot, 256), 256, 0, stream>>>(X, Y, tot);
  }

  // --- head: conv2 (K=3, 96->128) straight into d_out [B, N3, 128] ---
  cheb((float*)d_out, X, conv2_t, conv2_b, 3, B, N3, 96, 96, 128, 128, g3, T1, T2, PK, stream);
}
